// SelfContraction_17600775979386
// MI455X (gfx1250) — matrix operand order repaired and checked by reading, not compiled, not run on hardware
//
#include <hip/hip_runtime.h>

typedef _Float16 half8  __attribute__((ext_vector_type(8)));
typedef _Float16 half16 __attribute__((ext_vector_type(16)));
typedef float    float8 __attribute__((ext_vector_type(8)));

#define CC   128
#define EE   10
#define KTOT 896      // 7 * 128
#define FOUT 512      // 4 * 128 (output row width)
#define TM   32       // nodes per tile (2 WMMA row-subtiles)

#define D_INV3   0.57735026918962576451f
#define NORM_C_  0.08838834764831843f   // 1/sqrt(128)
#define SC0      0.08838834764831843f   // 1/sqrt(C)
#define SC1      0.0625f                // 1/sqrt(2C)
#define SC2      0.04419417382415922f   // 1/sqrt(4C)

// ---------------------------------------------------------------- bucketing
__global__ void k_init(int* counts, int* cursor) {
    int t = threadIdx.x;
    if (t < EE) { counts[t] = 0; cursor[t] = 0; }
}

__global__ void k_classify(const float* __restrict__ attrs, int* __restrict__ elemId,
                           int* __restrict__ counts, int n) {
    int i = blockIdx.x * blockDim.x + threadIdx.x;
    if (i >= n) return;
    int e = 0;
    #pragma unroll
    for (int j = 0; j < EE; ++j)
        if (attrs[(size_t)i * EE + j] > 0.5f) e = j;
    elemId[i] = e;
    atomicAdd(&counts[e], 1);
}

__global__ void k_prefix(const int* __restrict__ counts, int* __restrict__ ab) {
    if (threadIdx.x == 0 && blockIdx.x == 0) {
        int base = 0;
        for (int e = 0; e < EE; ++e) {
            ab[e] = base;
            base += ((counts[e] + TM - 1) / TM) * TM;   // align buckets to 32
        }
        ab[EE] = base;
    }
}

__global__ void k_fill(int* __restrict__ perm, int* __restrict__ tileElem,
                       const int* __restrict__ ab, int padn) {
    int s = blockIdx.x * blockDim.x + threadIdx.x;
    if (s >= padn) return;
    perm[s] = -1;
    if ((s % TM) == 0) {
        int total = ab[EE];
        int te = -1;
        if (s < total) {
            for (int e = 0; e < EE; ++e)
                if (s >= ab[e] && s < ab[e + 1]) te = e;
        }
        tileElem[s / TM] = te;
    }
}

__global__ void k_scatter(const int* __restrict__ elemId, const int* __restrict__ ab,
                          int* __restrict__ cursor, int* __restrict__ perm, int n) {
    int i = blockIdx.x * blockDim.x + threadIdx.x;
    if (i >= n) return;
    int e = elemId[i];
    int pos = ab[e] + atomicAdd(&cursor[e], 1);
    perm[pos] = i;
}

// ------------------------------------------------- fold weights (fp32 -> f16)
// WfoldT[st][e][c][k] = sum_j Wcat_st[e][k][j] * lin_st[j][c] * NORM_C
__global__ __launch_bounds__(128) void k_fold(
    const float* __restrict__ c0w0, const float* __restrict__ c0w1,
    const float* __restrict__ c1w0, const float* __restrict__ c1w1,
    const float* __restrict__ c2w0, const float* __restrict__ c2w1,
    const float* __restrict__ lin0, const float* __restrict__ lin1,
    _Float16* __restrict__ Wf0, _Float16* __restrict__ Wf1)
{
    __shared__ float row[CC];
    int bid = blockIdx.x;
    int e   = bid / (2 * KTOT);
    int rem = bid % (2 * KTOT);
    int st  = rem / KTOT;
    int k   = rem % KTOT;
    int f   = k >> 7;
    int c   = k & 127;

    const float* src;
    float scale;
    if (f == 0)       { src = (st ? c0w1 : c0w0) + ((size_t)e * CC        + c) * CC; scale = SC0; }
    else if (f <= 2)  { src = (st ? c1w1 : c1w0) + ((size_t)e * 2 * CC + (f - 1) * CC + c) * CC; scale = SC1; }
    else              { src = (st ? c2w1 : c2w0) + ((size_t)e * 4 * CC + (f - 3) * CC + c) * CC; scale = SC2; }

    int t = threadIdx.x;
    row[t] = src[t] * scale;
    __syncthreads();

    const float* lin = st ? lin1 : lin0;
    float acc = 0.f;
    #pragma unroll 8
    for (int j = 0; j < CC; ++j)
        acc += row[j] * lin[(size_t)j * CC + t];

    _Float16* W = st ? Wf1 : Wf0;
    W[((size_t)e * CC + t) * KTOT + k] = (_Float16)(acc * NORM_C_);
}

// ---------------------------------------------------------------- main GEMM
// Block = 256 threads = 8 waves; one tile of 32 same-element nodes.
// Wave j owns output columns 16j..16j+15 and reuses one (Wfold0, Wfold1)
// B-fragment pair across 4 streams x 2 node-subtiles = 8 WMMAs per K-chunk.
// Next K-chunk's B is prefetched into registers before the WMMA chain.
__global__ __launch_bounds__(256) void k_main(
    const float* __restrict__ nf,  const float* __restrict__ sc,
    const float* __restrict__ tp2, const float* __restrict__ tp3,
    const int* __restrict__ perm,  const int* __restrict__ tileElem,
    const _Float16* __restrict__ Wf0, const _Float16* __restrict__ Wf1,
    float* __restrict__ out)
{
    extern __shared__ char smem[];
    _Float16* Xs   = (_Float16*)smem;                     // [4][32][896] f16
    int*      idxs = (int*)(smem + 4 * TM * KTOT * 2);    // [32]

    const int tile = blockIdx.x;
    const int e = tileElem[tile];
    if (e < 0) return;
    const int t = threadIdx.x;
    if (t < TM) idxs[t] = perm[tile * TM + t];
    __syncthreads();

    // ---- phase A: build X features (4 streams x 32 nodes x 896) in f16 LDS
    #pragma unroll
    for (int i = 0; i < 16; ++i) {
        int p = t + i * 256;            // 4096 (node, channel) pairs
        int node = p >> 7;
        int c    = p & 127;
        int idx  = idxs[node];
        float s = 0.f, vx = 0.f, vy = 0.f, vz = 0.f;
        if (idx >= 0) {
            const float* fp = nf + (size_t)idx * FOUT;
            s  = fp[c];
            vx = fp[CC + 3 * c + 0];
            vy = fp[CC + 3 * c + 1];
            vz = fp[CC + 3 * c + 2];
        }
        float w20 = tp2[c], w21 = tp2[CC + c], w22 = tp2[2 * CC + c], w23 = tp2[3 * CC + c];
        float m0a  = w20 * s * s;
        float svi  = s * D_INV3;
        float m1ax = w21 * svi * vx, m1ay = w21 * svi * vy, m1az = w21 * svi * vz;
        float m1bx = w22 * svi * vx, m1by = w22 * svi * vy, m1bz = w22 * svi * vz;
        float m0b  = w23 * (vx * vx + vy * vy + vz * vz) * D_INV3;
        float da   = (m1ax * vx + m1ay * vy + m1az * vz) * D_INV3;
        float db   = (m1bx * vx + m1by * vy + m1bz * vz) * D_INV3;

        _Float16* X0 = Xs + (size_t)(0 * TM + node) * KTOT + c;
        X0[0]       = (_Float16)s;
        X0[1 * CC]  = (_Float16)m0a;
        X0[2 * CC]  = (_Float16)m0b;
        X0[3 * CC]  = (_Float16)(tp3[0 * CC + c] * m0a * s);
        X0[4 * CC]  = (_Float16)(tp3[3 * CC + c] * da);
        X0[5 * CC]  = (_Float16)(tp3[5 * CC + c] * db);
        X0[6 * CC]  = (_Float16)(tp3[6 * CC + c] * m0b * s);

        float w31 = tp3[1 * CC + c] * D_INV3;
        float w32 = tp3[2 * CC + c] * D_INV3;
        float w34 = tp3[4 * CC + c] * D_INV3;
        float w37 = tp3[7 * CC + c] * D_INV3;
        float vm[3]  = { vx,  vy,  vz  };
        float mam[3] = { m1ax, m1ay, m1az };
        float mbm[3] = { m1bx, m1by, m1bz };
        #pragma unroll
        for (int m = 0; m < 3; ++m) {
            _Float16* X1 = Xs + (size_t)((m + 1) * TM + node) * KTOT + c;
            X1[0]      = (_Float16)vm[m];
            X1[1 * CC] = (_Float16)mam[m];
            X1[2 * CC] = (_Float16)mbm[m];
            X1[3 * CC] = (_Float16)(w31 * m0a * vm[m]);
            X1[4 * CC] = (_Float16)(w32 * mam[m] * s);
            X1[5 * CC] = (_Float16)(w34 * mbm[m] * s);
            X1[6 * CC] = (_Float16)(w37 * m0b * vm[m]);
        }
    }
    __syncthreads();

    // ---- phase B: WMMA K-loop with register-pipelined B fragments
    const int wave = t >> 5;         // wave j -> columns 16j..16j+15
    const int lane = t & 31;
    const int ln   = lane & 15;
    const int hiL  = lane >> 4;      // 0: lanes 0-15, 1: lanes 16-31
    const int col  = wave * 16 + ln;

    // B layout: lane = col N; K split {0-7,16-23} / {8-15,24-31} (weights stored transposed)
    const _Float16* pB0 = Wf0 + (size_t)e * (CC * KTOT) + (size_t)col * KTOT + hiL * 8;
    const _Float16* pB1 = Wf1 + (size_t)e * (CC * KTOT) + (size_t)col * KTOT + hiL * 8;
    // A layout: lane = row M; K split {0-7,16-23} / {8-15,24-31}
    const _Float16* Abase = Xs + (size_t)ln * KTOT + hiL * 8;

    auto loadB = [](const _Float16* p) -> half16 {
        half8 lo = *(const half8*)(p);
        half8 hi = *(const half8*)(p + 16);
        half16 r;
        #pragma unroll
        for (int q = 0; q < 8; ++q) { r[q] = lo[q]; r[8 + q] = hi[q]; }
        return r;
    };
    auto loadA = [](const _Float16* p) -> half16 {
        half8 lo = *(const half8*)(p);
        half8 hi = *(const half8*)(p + 16);
        half16 r;
        #pragma unroll
        for (int q = 0; q < 8; ++q) { r[q] = lo[q]; r[8 + q] = hi[q]; }
        return r;
    };

    float8 acc[8] = { float8{}, float8{}, float8{}, float8{},
                      float8{}, float8{}, float8{}, float8{} };

    half16 b0 = loadB(pB0);
    half16 b1 = loadB(pB1);

    for (int kc = 0; kc < 28; ++kc) {
        const int k0 = kc * 32;
        const int kn = (kc < 27) ? (k0 + 32) : k0;   // branch-free prefetch
        half16 nb0 = loadB(pB0 + kn);
        half16 nb1 = loadB(pB1 + kn);
        #pragma unroll
        for (int sm = 0; sm < 4; ++sm) {
            #pragma unroll
            for (int u = 0; u < 2; ++u) {
                const _Float16* ap = Abase + (size_t)(sm * TM + u * 16) * KTOT + k0;
                half16 a = loadA(ap);
                acc[sm * 2 + u] = __builtin_amdgcn_wmma_f32_16x16x32_f16(
                    false, a, false, (sm == 0) ? b0 : b1, (short)0,
                    acc[sm * 2 + u], false, false);
            }
        }
        b0 = nb0;
        b1 = nb1;
    }

    // ---- phase C: scatter D (16x16 f32 layout) + skip connection
    #pragma unroll
    for (int sm = 0; sm < 4; ++sm) {
        #pragma unroll
        for (int u = 0; u < 2; ++u) {
            const float8 a = acc[sm * 2 + u];
            #pragma unroll
            for (int r = 0; r < 8; ++r) {
                int M   = r + hiL * 8;
                int idx = idxs[u * 16 + M];
                if (idx >= 0) {
                    int oc = (sm == 0) ? col : (CC + col * 3 + (sm - 1));
                    size_t off = (size_t)idx * FOUT + oc;
                    out[off] = a[r] + sc[off];
                }
            }
        }
    }
}

// ---------------------------------------------------------------- launch
extern "C" void kernel_launch(void* const* d_in, const int* in_sizes, int n_in,
                              void* d_out, int out_size, void* d_ws, size_t ws_size,
                              hipStream_t stream) {
    const float* nf    = (const float*)d_in[0];
    const float* attrs = (const float*)d_in[1];
    const float* sc    = (const float*)d_in[2];
    const float* tp2   = (const float*)d_in[3];
    const float* tp3   = (const float*)d_in[4];
    const float* c0w0  = (const float*)d_in[5];
    const float* c0w1  = (const float*)d_in[6];
    const float* c1w0  = (const float*)d_in[7];
    const float* c1w1  = (const float*)d_in[8];
    const float* c2w0  = (const float*)d_in[9];
    const float* c2w1  = (const float*)d_in[10];
    const float* lin0  = (const float*)d_in[11];
    const float* lin1  = (const float*)d_in[12];
    float* out = (float*)d_out;

    const int N    = in_sizes[0] / FOUT;
    const int PADN = ((N + TM - 1) / TM) * TM + TM * EE;   // buckets padded to 32
    const int nTiles = PADN / TM;

    auto AL = [](size_t x) { return (x + 255) & ~(size_t)255; };
    char* ws = (char*)d_ws;
    size_t off = 0;
    int* counts   = (int*)(ws + off); off = AL(off + EE * 4);
    int* cursor   = (int*)(ws + off); off = AL(off + EE * 4);
    int* ab       = (int*)(ws + off); off = AL(off + (EE + 1) * 4);
    int* elemId   = (int*)(ws + off); off = AL(off + (size_t)N * 4);
    int* perm     = (int*)(ws + off); off = AL(off + (size_t)PADN * 4);
    int* tileElem = (int*)(ws + off); off = AL(off + (size_t)nTiles * 4);
    _Float16* Wf0 = (_Float16*)(ws + off); off = AL(off + (size_t)EE * CC * KTOT * 2);
    _Float16* Wf1 = (_Float16*)(ws + off); off = AL(off + (size_t)EE * CC * KTOT * 2);
    (void)ws_size; (void)n_in; (void)out_size;

    k_init<<<1, 64, 0, stream>>>(counts, cursor);
    k_classify<<<(N + 255) / 256, 256, 0, stream>>>(attrs, elemId, counts, N);
    k_prefix<<<1, 32, 0, stream>>>(counts, ab);
    k_fill<<<(PADN + 255) / 256, 256, 0, stream>>>(perm, tileElem, ab, PADN);
    k_scatter<<<(N + 255) / 256, 256, 0, stream>>>(elemId, ab, cursor, perm, N);
    k_fold<<<EE * 2 * KTOT, 128, 0, stream>>>(c0w0, c0w1, c1w0, c1w1, c2w0, c2w1,
                                              lin0, lin1, Wf0, Wf1);
    const size_t smem = (size_t)4 * TM * KTOT * 2 + 128;   // X features + idx table
    k_main<<<nTiles, 256, smem, stream>>>(nf, sc, tp2, tp3, perm, tileElem,
                                          Wf0, Wf1, out);
}
